// Tracker_58884001628710
// MI455X (gfx1250) — matrix operand order repaired and checked by reading, not compiled, not run on hardware
//
#include <hip/hip_runtime.h>
#include <hip/hip_bf16.h>

// ---------------------------------------------------------------------------
// MI455X (gfx1250) fused LSTM-cell tracker step.
// Shapes: B=8192, L=S=32, DP=512, M=256, T=256.
// Effective math (h0=c0=0): gates[i,j,o] = gather(x:8192x768) @ W'(768x768)
// then c = sigmoid(i)*tanh(j); h = tanh(c)*sigmoid(o); logits = c @ Wt + bt.
// Path: bf16 WMMA (v_wmma_f32_16x16x32_bf16), fp32 accumulate.
//
// Round-3 change: all 6 B fragments of a K-step are loaded into distinct
// registers before the 12 WMMAs, so the loads form one clause with partial
// loadcnt waits instead of the load->wait(0)->wmma serialization seen in
// round-2 asm. ~176 VGPRs live, still no spill risk for wave32.
// ---------------------------------------------------------------------------

typedef __bf16 bf16;
typedef __attribute__((ext_vector_type(16))) __bf16 v16bf;
typedef __attribute__((ext_vector_type(8)))  __bf16 v8bf;
typedef __attribute__((ext_vector_type(8)))  float  v8f;

union FragAB { v16bf full; v8bf half[2]; };

__device__ __forceinline__ bf16 f2bf(float f) {
  // round-to-nearest-even f32 -> bf16
  unsigned u = __builtin_bit_cast(unsigned, f);
  unsigned r = u + 0x7FFFu + ((u >> 16) & 1u);
  return __builtin_bit_cast(bf16, (unsigned short)(r >> 16));
}

__device__ __forceinline__ float sigmoidf_(float x) {
  return 1.0f / (1.0f + __expf(-x));
}
__device__ __forceinline__ float tanhf_(float x) {
  return 2.0f / (1.0f + __expf(-2.0f * x)) - 1.0f;
}

// ---------------------------------------------------------------------------
// Gather the three length-indexed 256-wide slices per batch row and convert
// to bf16: xb[b][0:256]=bufs[b,blen-1,:256], [256:512]=stacks[b,slen-1,:256],
// [512:768]=stacks[b,slen-2,:256]. Only ~24 MB of the 1 GB inputs is touched.
// ---------------------------------------------------------------------------
__global__ __launch_bounds__(256) void prep_x_kernel(
    const float* __restrict__ bufs, const float* __restrict__ stacks,
    const int* __restrict__ blen, const int* __restrict__ slen,
    bf16* __restrict__ xb, int total)
{
  int i = blockIdx.x * 256 + threadIdx.x;
  if (i >= total) return;
  int b = i / 768;
  int k = i - b * 768;
  float v;
  if (k < 256) {
    int l = blen[b] - 1;
    v = bufs[((size_t)b * 32 + l) * 512 + k];
  } else if (k < 512) {
    int l = slen[b] - 1;
    v = stacks[((size_t)b * 32 + l) * 512 + (k - 256)];
  } else {
    int l = slen[b] - 2;
    v = stacks[((size_t)b * 32 + l) * 512 + (k - 512)];
  }
  xb[i] = f2bf(v);
}

// ---------------------------------------------------------------------------
// Build WT[m][k] (768x768 bf16), m = gate-major output column (g in {i,j,o},
// t in [0,256)), k = reduction index. Transposed so B-fragments are
// contiguous along K (matches CDNA5 16-bit B-matrix VGPR layout loads).
// f-gate columns of W are dead (c0==0) and are dropped here.
// ---------------------------------------------------------------------------
__global__ __launch_bounds__(256) void prep_w_kernel(
    const float* __restrict__ W, bf16* __restrict__ wtb, int total)
{
  int i = blockIdx.x * 256 + threadIdx.x;
  if (i >= total) return;
  int m = i / 768;          // output column in pruned {i,j,o} ordering
  int k = i - m * 768;
  int g = m >> 8;           // 0=i, 1=j, 2=o
  int t = m & 255;
  int nsrc = (g == 2) ? (768 + t) : (g * 256 + t);  // skip f block (512..767)
  wtb[i] = f2bf(W[(size_t)k * 1024 + nsrc]);
}

// ---------------------------------------------------------------------------
// Main fused GEMM + activations.
// Grid: (128, 2) blocks x 256 threads (8 wave32).
// Block tile: 64 rows x 128 t-cols (x 3 gates).
// Wave (rh = w&1, th = w>>1): rows [rowb, rowb+32), t-cols [tbase, tbase+32)
//   => 12 accumulators: [rt in 0..1][tt in 0..1][g in 0..2] of 16x16 f32.
// Per K-step (32): 2 A fragments (4 x b128), 6 B fragments (12 x b128, all
// live simultaneously), 12 WMMAs -- each B fragment feeds both row-tiles.
// A fragment (16x32 bf16): lane l<16 -> row l, k {0..7, 16..23};
//                          lane l>=16 -> row l-16, k {8..15, 24..31}.
// B fragment (32x16 bf16): lane l<16 -> col l, k {0..7, 16..23}; l>=16 -> k {8..15, 24..31}.
// C layout: VGPR r -> row r + 8*(lane>=16), col = lane&15.
// ---------------------------------------------------------------------------
__global__ __launch_bounds__(256) void fused_lstm_gemm(
    const bf16* __restrict__ xb, const bf16* __restrict__ wtb,
    const float* __restrict__ bvec, float* __restrict__ out)
{
  const int tid  = threadIdx.x;
  const int wave = tid >> 5;
  const int lane = tid & 31;
  const int rh = wave & 1;                     // row half of block (32 rows)
  const int th = wave >> 1;                    // t sub-block (32 t-cols)
  const int rowb  = blockIdx.x * 64 + rh * 32; // wave's first row
  const int tbase = blockIdx.y * 128 + th * 32;// wave's first t col
  const int lhi = lane >> 4;                   // lane half
  const int llo = lane & 15;

  v8f acc[12];                                 // idx = rt*6 + (tt*3 + g)
#pragma unroll
  for (int i = 0; i < 12; ++i) acc[i] = (v8f){0.f,0.f,0.f,0.f,0.f,0.f,0.f,0.f};

  const bf16* arow0 = xb + (size_t)(rowb + llo) * 768 + 8 * lhi;
  const bf16* arow1 = arow0 + (size_t)16 * 768;
  const bf16* bbase = wtb + (size_t)(tbase + llo) * 768 + 8 * lhi;

  for (int ks = 0; ks < 24; ++ks) {
    const int kb = ks * 32;

    // ---- load phase: 2 A frags + 6 B frags, all into distinct registers ---
    FragAB a0, a1;
    a0.half[0] = *(const v8bf*)(arow0 + kb);        // k {0..7} / {8..15}
    a0.half[1] = *(const v8bf*)(arow0 + kb + 16);   // k {16..23} / {24..31}
    a1.half[0] = *(const v8bf*)(arow1 + kb);
    a1.half[1] = *(const v8bf*)(arow1 + kb + 16);

    FragAB bm[6];                                   // idx = tt*3 + g
#pragma unroll
    for (int tt = 0; tt < 2; ++tt) {
#pragma unroll
      for (int g = 0; g < 3; ++g) {
        const bf16* bp = bbase + (size_t)(g * 256 + tt * 16) * 768 + kb;
        bm[tt * 3 + g].half[0] = *(const v8bf*)(bp);      // k {0..7} / {8..15}
        bm[tt * 3 + g].half[1] = *(const v8bf*)(bp + 16);
      }
    }
    if (ks + 1 < 24) {
      __builtin_prefetch(arow0 + kb + 32, 0, 0);
      __builtin_prefetch(arow1 + kb + 32, 0, 0);
    }

    // ---- math phase: 12 back-to-back WMMAs ------------------------------
#pragma unroll
    for (int i = 0; i < 6; ++i) {
      acc[i] = __builtin_amdgcn_wmma_f32_16x16x32_bf16(
          false, a0.full, false, bm[i].full, (short)0, acc[i], false, false);
      acc[6 + i] = __builtin_amdgcn_wmma_f32_16x16x32_bf16(
          false, a1.full, false, bm[i].full, (short)0, acc[6 + i], false, false);
    }
  }

  // Epilogue: bias + LSTM nonlinearity, store state_out = [c | h] (stride 512)
#pragma unroll
  for (int rt = 0; rt < 2; ++rt) {
    const int rowtop = rowb + rt * 16 + lhi * 8;
#pragma unroll
    for (int tt = 0; tt < 2; ++tt) {
      const int ncol = tbase + tt * 16 + llo;   // t index in [0,256)
      const float bi = bvec[ncol];
      const float bj = bvec[256 + ncol];
      const float bo = bvec[768 + ncol];
      v8f gi = acc[rt * 6 + tt * 3 + 0];
      v8f gj = acc[rt * 6 + tt * 3 + 1];
      v8f go = acc[rt * 6 + tt * 3 + 2];
#pragma unroll
      for (int r = 0; r < 8; ++r) {
        float c = sigmoidf_(gi[r] + bi) * tanhf_(gj[r] + bj);
        float h = tanhf_(c) * sigmoidf_(go[r] + bo);
        float* rp = out + (size_t)(rowtop + r) * 512;
        rp[ncol]       = c;
        rp[256 + ncol] = h;
      }
    }
  }
}

// ---------------------------------------------------------------------------
// logits[b, 0:4] = new_c[b, 0:256] @ Wt(256x4) + bt.  One wave32 per row,
// coalesced loads, xor-shuffle reduction. Memory-bound (~8 MB read).
// ---------------------------------------------------------------------------
__global__ __launch_bounds__(256) void logits_kernel(
    const float* __restrict__ state, const float* __restrict__ Wt,
    const float* __restrict__ bt, float* __restrict__ logits)
{
  const int lane = threadIdx.x & 31;
  const int row  = blockIdx.x * 8 + (threadIdx.x >> 5);
  const float* crow = state + (size_t)row * 512;

  float p0 = 0.f, p1 = 0.f, p2 = 0.f, p3 = 0.f;
#pragma unroll
  for (int kk = 0; kk < 8; ++kk) {
    int t = lane + kk * 32;
    float c = crow[t];
    const float* w = Wt + t * 4;
    p0 += c * w[0]; p1 += c * w[1]; p2 += c * w[2]; p3 += c * w[3];
  }
#pragma unroll
  for (int off = 16; off > 0; off >>= 1) {
    p0 += __shfl_xor(p0, off, 32);
    p1 += __shfl_xor(p1, off, 32);
    p2 += __shfl_xor(p2, off, 32);
    p3 += __shfl_xor(p3, off, 32);
  }
  if (lane == 0) {
    float* lp = logits + (size_t)row * 4;
    lp[0] = p0 + bt[0]; lp[1] = p1 + bt[1];
    lp[2] = p2 + bt[2]; lp[3] = p3 + bt[3];
  }
}

// ---------------------------------------------------------------------------
extern "C" void kernel_launch(void* const* d_in, const int* in_sizes, int n_in,
                              void* d_out, int out_size, void* d_ws, size_t ws_size,
                              hipStream_t stream) {
  const float* bufs   = (const float*)d_in[0];
  const float* stacks = (const float*)d_in[1];
  const float* W      = (const float*)d_in[2];
  const float* bvec   = (const float*)d_in[3];
  const float* Wt     = (const float*)d_in[4];
  const float* bt     = (const float*)d_in[5];
  const int*   blen   = (const int*)d_in[6];
  const int*   slen   = (const int*)d_in[7];

  float* out    = (float*)d_out;                    // state_out: 8192*512
  float* logits = out + (size_t)8192 * 512;         // logits:    8192*4

  bf16* xb  = (bf16*)d_ws;                          // 8192*768 bf16 (12.6 MB)
  bf16* wtb = xb + (size_t)8192 * 768;              // 768*768  bf16 ( 1.2 MB)

  const int NX = 8192 * 768;
  const int NW = 768 * 768;
  prep_x_kernel<<<(NX + 255) / 256, 256, 0, stream>>>(bufs, stacks, blen, slen, xb, NX);
  prep_w_kernel<<<(NW + 255) / 256, 256, 0, stream>>>(W, wtb, NW);
  fused_lstm_gemm<<<dim3(128, 2), 256, 0, stream>>>(xb, wtb, bvec, out);
  logits_kernel<<<1024, 256, 0, stream>>>(out, Wt, bt, logits);
}
